// QuantizedDecoder_2379411882288
// MI455X (gfx1250) — matrix operand order repaired and checked by reading, not compiled, not run on hardware
//
#include <hip/hip_runtime.h>
#include <hip/hip_bf16.h>

// ---------- sizes (fixed by the reference) ----------
#define BB 32768
#define DD 512
#define KK 512
#define HH 2048
#define OO 1024

typedef __attribute__((ext_vector_type(16))) __bf16 v16bf;
typedef __attribute__((ext_vector_type(8)))  float  v8f;
typedef __attribute__((ext_vector_type(4)))  unsigned int u32x4;
typedef __attribute__((ext_vector_type(4)))  int i32x4;

union frag16 { u32x4 u[2]; v16bf v; };

// ---------------------------------------------------------------------------
// CDNA5 async global->LDS copy (ASYNCcnt-tracked) with sync fallback
// ---------------------------------------------------------------------------
#if defined(__HIP_DEVICE_COMPILE__) && \
    __has_builtin(__builtin_amdgcn_global_load_async_to_lds_b128)
#define ASYNC_LDS 1
#endif

#ifdef ASYNC_LDS
typedef __attribute__((address_space(1))) i32x4 g_i32x4;
typedef __attribute__((address_space(3))) i32x4 l_i32x4;
#endif

__device__ __forceinline__ void async_cp16(const __bf16* g, __bf16* l) {
#ifdef ASYNC_LDS
    __builtin_amdgcn_global_load_async_to_lds_b128(
        (g_i32x4*)g, (l_i32x4*)l, 0, 0);
#else
    *(u32x4*)l = *(const u32x4*)g;
#endif
}

template <int N>
__device__ __forceinline__ void wait_async() {
#if defined(__HIP_DEVICE_COMPILE__)
#if __has_builtin(__builtin_amdgcn_s_wait_asynccnt)
    __builtin_amdgcn_s_wait_asynccnt(N);
#else
    asm volatile("s_wait_asynccnt %0" ::"n"(N) : "memory");
#endif
#endif
}

// ---------------------------------------------------------------------------
// Fragment loaders for v_wmma_f32_16x16x32_bf16 (wave32)
// A (16x32 MxK): lanes 0-15 row=lane, K k..k+7 & k+16..k+23;
//                lanes16-31 row=lane-16, K k+8..k+15 & k+24..k+31
// B (32x16 KxN) from K-contiguous storage: lanes 0-15 col=lane, K k..k+7 & k+16..k+23;
//                lanes16-31 col=lane-16, K k+8..k+15 & k+24..k+31
// ---------------------------------------------------------------------------
__device__ __forceinline__ v16bf load_a_frag(const __bf16* A, int lda, int row, int k, int lane) {
    const __bf16* p = A + (size_t)(row + (lane & 15)) * lda + k + ((lane >> 4) << 3);
    frag16 f;
    f.u[0] = *(const u32x4*)(p);
    f.u[1] = *(const u32x4*)(p + 16);
    return f.v;
}

#define LDSB 40  // padded row stride (elements): 80B, 16B-aligned, conflict-free

__device__ __forceinline__ v16bf lds_a_frag(const __bf16* base, int row, int lane) {
    const __bf16* p = base + (row + (lane & 15)) * LDSB + ((lane >> 4) << 3);
    frag16 f;
    f.u[0] = *(const u32x4*)(p);
    f.u[1] = *(const u32x4*)(p + 16);
    return f.v;
}

__device__ __forceinline__ v16bf lds_b_frag(const __bf16* base, int col, int lane) {
    const __bf16* p = base + (col + (lane & 15)) * LDSB + ((lane >> 4) << 3);
    frag16 f;
    f.u[0] = *(const u32x4*)(p);
    f.u[1] = *(const u32x4*)(p + 16);
    return f.v;
}

// ---------------------------------------------------------------------------
// f32 -> bf16 elementwise convert (4 elems/thread, n % 4 == 0)
// ---------------------------------------------------------------------------
__global__ void k_cvt_bf16(const float* __restrict__ x, __bf16* __restrict__ y, size_t n) {
    size_t i = ((size_t)blockIdx.x * blockDim.x + threadIdx.x) * 4;
    if (i + 3 < n) {
        float4 v = *(const float4*)(x + i);
        y[i + 0] = (__bf16)v.x;
        y[i + 1] = (__bf16)v.y;
        y[i + 2] = (__bf16)v.z;
        y[i + 3] = (__bf16)v.w;
    }
}

// ---------------------------------------------------------------------------
// Codebook prep: bf16 copy of emb + per-code squared norm. One block per code.
// ---------------------------------------------------------------------------
__global__ void k_emb_prep(const float* __restrict__ emb, __bf16* __restrict__ eb,
                           float* __restrict__ enorm) {
    int k = blockIdx.x;
    const float* row = emb + (size_t)k * DD;
    __bf16* orow = eb + (size_t)k * DD;
    float s = 0.f;
    for (int d = threadIdx.x; d < DD; d += blockDim.x) {
        float v = row[d];
        orow[d] = (__bf16)v;
        s += v * v;
    }
    for (int off = 16; off; off >>= 1) s += __shfl_down(s, off, 32);
    __shared__ float red[8];
    int lane = threadIdx.x & 31, w = threadIdx.x >> 5;
    if (lane == 0) red[w] = s;
    __syncthreads();
    if (threadIdx.x == 0) {
        float t = 0.f;
        int nw = blockDim.x >> 5;
        for (int i = 0; i < nw; i++) t += red[i];
        enorm[k] = t;
    }
}

// ---------------------------------------------------------------------------
// Transpose + convert weights: W [K x N] f32 row-major -> Wt [N x K] bf16.
// ---------------------------------------------------------------------------
__global__ void k_transpose_cvt(const float* __restrict__ W, __bf16* __restrict__ Wt,
                                int K, int N) {
    __shared__ __bf16 tile[32][33];
    int k0 = blockIdx.y * 32, n0 = blockIdx.x * 32;
    int tx = threadIdx.x & 31, ty = threadIdx.x >> 5;
    for (int r = ty; r < 32; r += 8)
        tile[r][tx] = (__bf16)W[(size_t)(k0 + r) * N + (n0 + tx)];
    __syncthreads();
    for (int r = ty; r < 32; r += 8)
        Wt[(size_t)(n0 + r) * K + (k0 + tx)] = tile[tx][r];
}

// ---------------------------------------------------------------------------
// VQ argmin: dist = ||e||^2 - 2 z.e via WMMA, codebook slabs async-staged in
// LDS (double buffered), z rows resident in registers.
// 8 waves/block, 16 rows/wave -> 128 rows/block.
// ---------------------------------------------------------------------------
#define ESTRIDE 520  // 512 + 8 pad: 1040B rows, 16B aligned, conflict-free

__global__ __launch_bounds__(256)
void k_vq_argmin(const __bf16* __restrict__ zb, const __bf16* __restrict__ eb,
                 const float* __restrict__ enorm, int* __restrict__ idx) {
    __shared__ __bf16 Es[2][16 * ESTRIDE];

    int lane = threadIdx.x & 31;
    int w    = threadIdx.x >> 5;
    int t    = threadIdx.x;
    int row  = blockIdx.x * 128 + w * 16;

    // preload this wave's 16 z rows (full D) into registers: 16 frags
    v16bf afr[16];
#pragma unroll
    for (int kt = 0; kt < 16; ++kt)
        afr[kt] = load_a_frag(zb, DD, row, kt * 32, lane);

    // stage one 16-code x 512 slab: 16KB = 1024 x 16B chunks, 4 per thread
    auto stage = [&](int buf, int code0) {
#pragma unroll
        for (int it = 0; it < 4; ++it) {
            int c = t + it * 256;          // 0..1023
            int r = c >> 6;                // code row 0..15
            int ch = (c & 63) * 8;         // element offset 0..504
            async_cp16(eb + (size_t)(code0 + r) * DD + ch,
                       &Es[buf][r * ESTRIDE + ch]);
        }
    };

    float minv[8];
    int   mini[8];
#pragma unroll
    for (int r = 0; r < 8; r++) { minv[r] = 3.4e38f; mini[r] = 0; }

    stage(0, 0);
    for (int ct = 0; ct < KK / 16; ++ct) {
#ifdef ASYNC_LDS
        if (ct + 1 < KK / 16) { stage((ct + 1) & 1, (ct + 1) * 16); wait_async<4>(); }
        else                  { wait_async<0>(); }
#else
        if (ct + 1 < KK / 16) stage((ct + 1) & 1, (ct + 1) * 16);
#endif
        __syncthreads();

        const __bf16* Eb = Es[ct & 1];
        v8f acc = {};
#pragma unroll
        for (int kt = 0; kt < 16; ++kt) {
            const __bf16* p = Eb + (lane & 15) * ESTRIDE + kt * 32 + ((lane >> 4) << 3);
            frag16 f;
            f.u[0] = *(const u32x4*)(p);
            f.u[1] = *(const u32x4*)(p + 16);
            acc = __builtin_amdgcn_wmma_f32_16x16x32_bf16(false, afr[kt], false, f.v,
                                                          (short)0, acc, false, false);
        }
        int n = ct * 16 + (lane & 15);
        float en = enorm[n];
#pragma unroll
        for (int r = 0; r < 8; r++) {
            float dist = en - 2.0f * acc[r];
            if (dist < minv[r]) { minv[r] = dist; mini[r] = n; }
        }
        __syncthreads();
    }

    // reduce across the 16 lanes sharing each row block (xor within half-wave)
#pragma unroll
    for (int off = 1; off < 16; off <<= 1) {
#pragma unroll
        for (int r = 0; r < 8; r++) {
            float ov = __shfl_xor(minv[r], off, 32);
            int   oi = __shfl_xor(mini[r], off, 32);
            if (ov < minv[r] || (ov == minv[r] && oi < mini[r])) {
                minv[r] = ov; mini[r] = oi;
            }
        }
    }
    if ((lane & 15) == 0) {
        int mbase = row + 8 * (lane >> 4);
#pragma unroll
        for (int r = 0; r < 8; r++) idx[mbase + r] = mini[r];
    }
}

// ---------------------------------------------------------------------------
// Gather quantized rows (f32 + bf16), accumulate squared error, emit index.
// ---------------------------------------------------------------------------
__global__ void k_gather(const float* __restrict__ z, const float* __restrict__ emb,
                         const int* __restrict__ idx,
                         float* __restrict__ q_out, __bf16* __restrict__ qb,
                         float* __restrict__ idx_out, float* __restrict__ accum) {
    int b = blockIdx.x;
    int k = idx[b];
    const float* e  = emb + (size_t)k * DD;
    const float* zr = z   + (size_t)b * DD;
    float s = 0.f;
    for (int d = threadIdx.x; d < DD; d += blockDim.x) {
        float q = e[d];
        float diff = zr[d] - q;
        s += diff * diff;
        q_out[(size_t)b * DD + d] = q;
        qb[(size_t)b * DD + d]    = (__bf16)q;
    }
    for (int off = 16; off; off >>= 1) s += __shfl_down(s, off, 32);
    __shared__ float red[8];
    int lane = threadIdx.x & 31, w = threadIdx.x >> 5;
    if (lane == 0) red[w] = s;
    __syncthreads();
    if (threadIdx.x == 0) {
        float t = 0.f;
        int nw = blockDim.x >> 5;
        for (int i = 0; i < nw; i++) t += red[i];
        atomicAdd(accum, t);
        idx_out[b] = (float)k;
    }
}

__global__ void k_loss(const float* __restrict__ accum, float* __restrict__ loss) {
    loss[0] = 1.25f * accum[0] / (float)((size_t)BB * DD);
}

// ---------------------------------------------------------------------------
// bf16 WMMA GEMM with async-LDS double-buffered 128x128x32 block tiles.
// 8 waves (2 M x 4 N), wave tile 64x32.
// ---------------------------------------------------------------------------
template <bool TANH>
__global__ __launch_bounds__(256)
void k_gemm_bf16(const __bf16* __restrict__ A, const __bf16* __restrict__ Bt,
                 const float* __restrict__ bias, void* __restrict__ Cout,
                 int N, int Kd, int out_is_bf16) {
    __shared__ __bf16 As[2][128 * LDSB];
    __shared__ __bf16 Bs[2][128 * LDSB];

    int lane = threadIdx.x & 31;
    int w    = threadIdx.x >> 5;
    int t    = threadIdx.x;
    int wm   = w & 1;              // 0..1
    int wn   = w >> 1;             // 0..3
    int bm0  = blockIdx.y * 128;
    int bn0  = blockIdx.x * 128;
    int wrow = wm * 64;
    int wcol = wn * 32;

    v8f acc[4][2] = {};

    // stage a 128x32 A tile + 128x32 B tile: 512 x 16B chunks each, 2/thread
    auto stage = [&](int buf, int k) {
#pragma unroll
        for (int it = 0; it < 2; ++it) {
            int c = t + it * 256;          // 0..511
            int r = c >> 2;                // row 0..127
            int ch = (c & 3) * 8;          // element offset 0/8/16/24
            async_cp16(A  + (size_t)(bm0 + r) * Kd + k + ch, &As[buf][r * LDSB + ch]);
            async_cp16(Bt + (size_t)(bn0 + r) * Kd + k + ch, &Bs[buf][r * LDSB + ch]);
        }
    };

    int nk = Kd >> 5;
    stage(0, 0);
    for (int kt = 0; kt < nk; ++kt) {
#ifdef ASYNC_LDS
        if (kt + 1 < nk) { stage((kt + 1) & 1, (kt + 1) << 5); wait_async<4>(); }
        else             { wait_async<0>(); }
#else
        if (kt + 1 < nk) stage((kt + 1) & 1, (kt + 1) << 5);
#endif
        __syncthreads();

        const __bf16* Ab = As[kt & 1] + wrow * LDSB;
        const __bf16* Bb = Bs[kt & 1] + wcol * LDSB;
        v16bf bf0 = lds_b_frag(Bb, 0, lane);
        v16bf bf1 = lds_b_frag(Bb, 16, lane);
#pragma unroll
        for (int i = 0; i < 4; i++) {
            v16bf af = lds_a_frag(Ab, 16 * i, lane);
            acc[i][0] = __builtin_amdgcn_wmma_f32_16x16x32_bf16(false, af, false, bf0,
                                                                (short)0, acc[i][0], false, false);
            acc[i][1] = __builtin_amdgcn_wmma_f32_16x16x32_bf16(false, af, false, bf1,
                                                                (short)0, acc[i][1], false, false);
        }
        __syncthreads();
    }

    int l15 = lane & 15, lh = lane >> 4;
    float*  Cf = (float*)Cout;
    __bf16* Cb = (__bf16*)Cout;
#pragma unroll
    for (int j = 0; j < 2; j++) {
        int n = bn0 + wcol + 16 * j + l15;
        float bv = bias[n];
#pragma unroll
        for (int i = 0; i < 4; i++) {
#pragma unroll
            for (int r = 0; r < 8; r++) {
                int m = bm0 + wrow + 16 * i + 8 * lh + r;
                float v = acc[i][j][r] + bv;
                if (TANH) v = tanhf(v);
                if (out_is_bf16) Cb[(size_t)m * N + n] = (__bf16)v;
                else             Cf[(size_t)m * N + n] = v;
            }
        }
    }
}

// ---------------------------------------------------------------------------
// launch
// ---------------------------------------------------------------------------
extern "C" void kernel_launch(void* const* d_in, const int* in_sizes, int n_in,
                              void* d_out, int out_size, void* d_ws, size_t ws_size,
                              hipStream_t stream) {
    // inputs: 0=state(unused) 1=z 2=emb 3=W1 4=b1 5=W2 6=b2 7=W3 8=b3
    const float* z   = (const float*)d_in[1];
    const float* emb = (const float*)d_in[2];
    const float* W1  = (const float*)d_in[3];
    const float* b1  = (const float*)d_in[4];
    const float* W2  = (const float*)d_in[5];
    const float* b2  = (const float*)d_in[6];
    const float* W3  = (const float*)d_in[7];
    const float* b3  = (const float*)d_in[8];

    // d_out: recon [B,O] | quantized_st [B,D] | vq_loss [1] | indices [B] (as f32)
    float* out_recon = (float*)d_out;
    float* out_quant = out_recon + (size_t)BB * OO;
    float* out_loss  = out_quant + (size_t)BB * DD;
    float* out_idx   = out_loss + 1;

    // workspace carve-up (all offsets 256B aligned; row strides 16B aligned)
    char* ws = (char*)d_ws;
    float*  accum = (float*)ws;                                   // 256 B
    __bf16* zb    = (__bf16*)(ws + 256);
    __bf16* qb    = zb + (size_t)BB * DD;
    __bf16* eb    = qb + (size_t)BB * DD;
    float*  enorm = (float*)(eb + (size_t)KK * DD);
    int*    idx   = (int*)(enorm + KK);
    __bf16* w1t   = (__bf16*)(idx + BB);
    __bf16* w2t   = w1t + (size_t)DD * HH;
    __bf16* w3t   = w2t + (size_t)HH * HH;
    __bf16* h1    = w3t + (size_t)HH * OO;
    __bf16* h2    = h1 + (size_t)BB * HH;

    (void)hipMemsetAsync(accum, 0, sizeof(float), stream);

    // prep: conversions / transposes / code norms
    {
        size_t n = (size_t)BB * DD;
        k_cvt_bf16<<<dim3((unsigned)(n / 4 / 256)), dim3(256), 0, stream>>>(z, zb, n);
    }
    k_emb_prep<<<dim3(KK), dim3(256), 0, stream>>>(emb, eb, enorm);
    k_transpose_cvt<<<dim3(HH / 32, DD / 32), dim3(256), 0, stream>>>(W1, w1t, DD, HH);
    k_transpose_cvt<<<dim3(HH / 32, HH / 32), dim3(256), 0, stream>>>(W2, w2t, HH, HH);
    k_transpose_cvt<<<dim3(OO / 32, HH / 32), dim3(256), 0, stream>>>(W3, w3t, HH, OO);

    // VQ: argmin over codebook, then gather + loss
    k_vq_argmin<<<dim3(BB / 128), dim3(256), 0, stream>>>(zb, eb, enorm, idx);
    k_gather<<<dim3(BB), dim3(256), 0, stream>>>(z, emb, idx, out_quant, qb, out_idx, accum);
    k_loss<<<dim3(1), dim3(1), 0, stream>>>(accum, out_loss);

    // MLP: q -> h1 -> h2 -> recon
    k_gemm_bf16<true><<<dim3(HH / 128, BB / 128), dim3(256), 0, stream>>>(
        qb, w1t, b1, (void*)h1, HH, DD, /*bf16 out*/1);
    k_gemm_bf16<true><<<dim3(HH / 128, BB / 128), dim3(256), 0, stream>>>(
        h1, w2t, b2, (void*)h2, HH, HH, 1);
    k_gemm_bf16<false><<<dim3(OO / 128, BB / 128), dim3(256), 0, stream>>>(
        h2, w3t, b3, (void*)out_recon, OO, HH, /*f32 out*/0);
}
